// GatedAttention_11922829214178
// MI455X (gfx1250) — compile-verified, matrix operand order checked by reading, not run on hardware
//
#include <hip/hip_runtime.h>
#include <math.h>

#define H_DIM 1024
#define NUM_HEADS 16
#define HEAD_DIM 64
#define SEQ 2048
#define BATCH 2
#define ROWS (BATCH*SEQ)     // 4096
#define E_QKV 3072
#define E_TOT 3088           // qkv (3072) + gate (16), both K=1024
#define E_PAD 3136           // padded to 49*64 so the GEMM K-loop is branch-free
#define EPS 1e-5f
#define PLD 40               // LDS row stride for P tiles (80B, 16B-aligned chunks)

typedef __bf16 bf16;
typedef __attribute__((ext_vector_type(16))) __bf16 v16bf;
typedef __attribute__((ext_vector_type(8)))  __bf16 v8bf;
typedef __attribute__((ext_vector_type(8)))  float  v8f;

union FragBF { v16bf v; v8bf h[2]; };

// A/B fragment for v_wmma_f32_16x16x32_bf16.
// Lane l: row/col = l&15, K-chunk base = (l>>4)*8; elems 0..7 = K+0..7, 8..15 = K+16..23.
// Requires K-contiguous storage with row stride `ldm` elements.
__device__ __forceinline__ v16bf load_frag(const bf16* tile, int ldm, int lane) {
  const int m  = lane & 15;
  const int hf = lane >> 4;
  const bf16* p = tile + (size_t)m * ldm + hf * 8;
  FragBF f;
  f.h[0] = *(const v8bf*)(p);
  f.h[1] = *(const v8bf*)(p + 16);
  return f.v;
}

__device__ __forceinline__ v8f wmma_bf16(v16bf a, v16bf b, v8f c) {
  return __builtin_amdgcn_wmma_f32_16x16x32_bf16(false, a, false, b, (short)0, c,
                                                 false, false);
}

// Double-buffered 16x64 GEMM tile: acc[t] += A(16xK) * W[t](16xK)^T, K = H_DIM.
__device__ __forceinline__ void gemm_16x64(const bf16* __restrict__ arow,
                                           const bf16* __restrict__ wrow,
                                           int lane, v8f acc[4]) {
  v16bf a  = load_frag(arow, H_DIM, lane);
  v16bf b0 = load_frag(wrow,                         H_DIM, lane);
  v16bf b1 = load_frag(wrow + (size_t)16 * H_DIM,    H_DIM, lane);
  v16bf b2 = load_frag(wrow + (size_t)32 * H_DIM,    H_DIM, lane);
  v16bf b3 = load_frag(wrow + (size_t)48 * H_DIM,    H_DIM, lane);
#pragma unroll 2
  for (int k = 0; k < H_DIM; k += 32) {
    const int kn = (k + 32) & (H_DIM - 1);   // last iter wraps: dead loads, no branch
    const v16bf an = load_frag(arow + kn, H_DIM, lane);
    const v16bf c0 = load_frag(wrow + kn, H_DIM, lane);
    const v16bf c1 = load_frag(wrow + (size_t)16 * H_DIM + kn, H_DIM, lane);
    const v16bf c2 = load_frag(wrow + (size_t)32 * H_DIM + kn, H_DIM, lane);
    const v16bf c3 = load_frag(wrow + (size_t)48 * H_DIM + kn, H_DIM, lane);
    acc[0] = wmma_bf16(a, b0, acc[0]);
    acc[1] = wmma_bf16(a, b1, acc[1]);
    acc[2] = wmma_bf16(a, b2, acc[2]);
    acc[3] = wmma_bf16(a, b3, acc[3]);
    a = an; b0 = c0; b1 = c1; b2 = c2; b3 = c3;
  }
}

// ---------------------------------------------------------------- kernel 1
// wcat[E_PAD][1024] = concat(qkv_w, gate_w, zeros) in bf16; owb = o_w in bf16.
__global__ void __launch_bounds__(256)
convert_weights_kernel(const float* __restrict__ qkvw, const float* __restrict__ gw,
                       const float* __restrict__ ow, bf16* __restrict__ wcat,
                       bf16* __restrict__ owb) {
  const size_t nqkv = (size_t)E_QKV * H_DIM;
  const size_t ncat = (size_t)E_TOT * H_DIM;
  const size_t npad = (size_t)E_PAD * H_DIM;
  const size_t tot  = npad + (size_t)H_DIM * H_DIM;
  for (size_t i = (size_t)blockIdx.x * blockDim.x + threadIdx.x; i < tot;
       i += (size_t)gridDim.x * blockDim.x) {
    if (i < nqkv)      wcat[i] = (bf16)qkvw[i];
    else if (i < ncat) wcat[i] = (bf16)gw[i - nqkv];
    else if (i < npad) wcat[i] = (bf16)0.0f;          // pad rows: zeros
    else               owb[i - npad] = (bf16)ow[i - npad];
  }
}

// ---------------------------------------------------------------- kernel 2
__global__ void __launch_bounds__(256)
prenorm_kernel(const float* __restrict__ x, const float* __restrict__ w,
               bf16* __restrict__ xn) {
  const int row = blockIdx.x;
  const float* xr = x + (size_t)row * H_DIM;
  float ss = 0.f;
  for (int i = threadIdx.x; i < H_DIM; i += 256) { float v = xr[i]; ss += v * v; }
  for (int off = 16; off > 0; off >>= 1) ss += __shfl_xor(ss, off, 32);
  __shared__ float red[8];
  if ((threadIdx.x & 31) == 0) red[threadIdx.x >> 5] = ss;
  __syncthreads();
  float sum = 0.f;
#pragma unroll
  for (int i = 0; i < 8; ++i) sum += red[i];
  const float rinv = rsqrtf(sum * (1.0f / H_DIM) + EPS);
  for (int i = threadIdx.x; i < H_DIM; i += 256)
    xn[(size_t)row * H_DIM + i] = (bf16)(xr[i] * rinv * w[i]);
}

// ---------------------------------------------------------------- kernel 3
// C[4096 x E_PAD] = xn * wcat^T ; scatter q/k (bf16), v (transposed bf16), gate (f32).
__global__ void __launch_bounds__(128)
gemm_qkvgate_kernel(const bf16* __restrict__ A, const bf16* __restrict__ W,
                    bf16* __restrict__ qk, bf16* __restrict__ vT,
                    float* __restrict__ gate) {
  const int wave = threadIdx.x >> 5;
  const int lane = threadIdx.x & 31;
  const int row0 = (blockIdx.y * 4 + wave) * 16;
  const int n0   = blockIdx.x * 64;
  v8f acc[4] = {};
  gemm_16x64(A + (size_t)row0 * H_DIM, W + (size_t)n0 * H_DIM, lane, acc);

  const int nn = lane & 15;
  const int hf = lane >> 4;
#pragma unroll
  for (int t = 0; t < 4; ++t) {
    const int n = n0 + t * 16 + nn;
    if (n >= E_TOT) continue;                        // drop pad columns
#pragma unroll
    for (int r = 0; r < 8; ++r) {
      const int m = row0 + r + 8 * hf;
      const float val = acc[t][r];
      if (n < 2 * H_DIM) {                           // q or k, row-major [4096][2048]
        qk[(size_t)m * (2 * H_DIM) + n] = (bf16)val;
      } else if (n < E_QKV) {                        // v -> vT[bh][feat][key]
        const int e = n - 2 * H_DIM;
        const int b_ = m / SEQ, s = m % SEQ;
        const int h = e >> 6, f = e & 63;
        vT[(((size_t)(b_ * NUM_HEADS + h)) * HEAD_DIM + f) * SEQ + s] = (bf16)val;
      } else {
        gate[(size_t)m * NUM_HEADS + (n - E_QKV)] = val;
      }
    }
  }
}

// ---------------------------------------------------------------- kernel 4
// In-place per-head RMS norm of q (cols 0..1023) and k (cols 1024..2047).
__global__ void __launch_bounds__(256)
qknorm_kernel(bf16* __restrict__ qk, const float* __restrict__ qw,
              const float* __restrict__ kw) {
  const int row = blockIdx.x;
  const int g  = threadIdx.x >> 3;   // 0..15 q heads, 16..31 k heads
  const int tl = threadIdx.x & 7;
  bf16* p = qk + (size_t)row * (2 * H_DIM) + g * HEAD_DIM + tl * 8;
  v8bf d = *(const v8bf*)p;
  float v[8], ss = 0.f;
#pragma unroll
  for (int i = 0; i < 8; ++i) { v[i] = (float)d[i]; ss += v[i] * v[i]; }
  for (int off = 4; off > 0; off >>= 1) ss += __shfl_xor(ss, off, 8);
  const float rinv = rsqrtf(ss * (1.0f / HEAD_DIM) + EPS);
  const float* wv = (g < NUM_HEADS) ? qw : kw;
  v8bf o;
#pragma unroll
  for (int i = 0; i < 8; ++i) o[i] = (bf16)(v[i] * rinv * wv[tl * 8 + i]);
  *(v8bf*)p = o;
}

// ---------------------------------------------------------------- kernel 5
// Flash attention: one wave per (b,h,16-query block); 32-key chunks.
__global__ void __launch_bounds__(128)
attn_kernel(const bf16* __restrict__ qk, const bf16* __restrict__ vT,
            const float* __restrict__ gate, bf16* __restrict__ attn) {
  __shared__ __align__(16) bf16 plds[4][16 * PLD];
  const int wave = threadIdx.x >> 5;
  const int lane = threadIdx.x & 31;
  const int wid  = blockIdx.x * 4 + wave;
  const int qb   = wid & 127;
  const int bh   = wid >> 7;            // 0..31
  const int b    = bh >> 4, h = bh & 15;
  const int q0   = qb * 16;

  const bf16* qbase = qk + ((size_t)(b * SEQ + q0)) * (2 * H_DIM) + h * HEAD_DIM;
  const bf16* kbase = qk + ((size_t)(b * SEQ)) * (2 * H_DIM) + H_DIM + h * HEAD_DIM;
  const bf16* vbase = vT + (size_t)bh * HEAD_DIM * SEQ;
  bf16* myp = &plds[wave][0];

  const v16bf qa0 = load_frag(qbase,      2 * H_DIM, lane);
  const v16bf qa1 = load_frag(qbase + 32, 2 * H_DIM, lane);

  v8f o[4] = {};
  float mrow[8], lrow[8];
#pragma unroll
  for (int r = 0; r < 8; ++r) { mrow[r] = -1e30f; lrow[r] = 0.f; }
  const float scale = 0.125f;  // 1/sqrt(64)
  const int nn = lane & 15;
  const int hfh = (lane >> 4) * 8;

  for (int kc = 0; kc < SEQ; kc += 32) {
    // Batch ALL loads for this chunk up-front: 4 K fragments + 4 V fragments.
    // The V loads overlap the softmax + LDS round-trip below.
    const v16bf kb0 = load_frag(kbase + (size_t)kc * (2 * H_DIM),             2 * H_DIM, lane);
    const v16bf kb1 = load_frag(kbase + (size_t)kc * (2 * H_DIM) + 32,        2 * H_DIM, lane);
    const v16bf kb2 = load_frag(kbase + (size_t)(kc + 16) * (2 * H_DIM),      2 * H_DIM, lane);
    const v16bf kb3 = load_frag(kbase + (size_t)(kc + 16) * (2 * H_DIM) + 32, 2 * H_DIM, lane);
    const v16bf vb0 = load_frag(vbase + (size_t)0  * SEQ + kc, SEQ, lane);
    const v16bf vb1 = load_frag(vbase + (size_t)16 * SEQ + kc, SEQ, lane);
    const v16bf vb2 = load_frag(vbase + (size_t)32 * SEQ + kc, SEQ, lane);
    const v16bf vb3 = load_frag(vbase + (size_t)48 * SEQ + kc, SEQ, lane);

    // scores: two 16x16 tiles over keys kc..kc+15, kc+16..kc+31 (K=64 -> 2 steps)
    v8f s0 = {}, s1 = {};
    s0 = wmma_bf16(qa0, kb0, s0);
    s0 = wmma_bf16(qa1, kb1, s0);
    s1 = wmma_bf16(qa0, kb2, s1);
    s1 = wmma_bf16(qa1, kb3, s1);

    // online softmax; rows live in VGPR index, keys across 16-lane halves
    float corr[8];
#pragma unroll
    for (int r = 0; r < 8; ++r) {
      float a0 = s0[r] * scale, a1 = s1[r] * scale;
      float mx = fmaxf(a0, a1);
      for (int off = 1; off < 16; off <<= 1) mx = fmaxf(mx, __shfl_xor(mx, off, 32));
      const float mnew = fmaxf(mrow[r], mx);
      const float p0 = __expf(a0 - mnew);
      const float p1 = __expf(a1 - mnew);
      s0[r] = p0; s1[r] = p1;
      float ps = p0 + p1;
      for (int off = 1; off < 16; off <<= 1) ps += __shfl_xor(ps, off, 32);
      corr[r] = __expf(mrow[r] - mnew);
      lrow[r] = lrow[r] * corr[r] + ps;
      mrow[r] = mnew;
    }
#pragma unroll
    for (int t = 0; t < 4; ++t)
#pragma unroll
      for (int r = 0; r < 8; ++r) o[t][r] *= corr[r];
    // C-layout -> A-layout via per-wave LDS (16 x 32 P tile)
#pragma unroll
    for (int r = 0; r < 8; ++r) {
      myp[(r + hfh) * PLD + nn]      = (bf16)s0[r];
      myp[(r + hfh) * PLD + 16 + nn] = (bf16)s1[r];
    }
    asm volatile("s_wait_dscnt 0x0" ::: "memory");
    const v16bf pa = load_frag(myp, PLD, lane);
    // P(16x32) x V(32x64): 4 N-tiles from transposed V (key-contiguous)
    o[0] = wmma_bf16(pa, vb0, o[0]);
    o[1] = wmma_bf16(pa, vb1, o[1]);
    o[2] = wmma_bf16(pa, vb2, o[2]);
    o[3] = wmma_bf16(pa, vb3, o[3]);
  }
  // epilogue: 1/l, sigmoid(gate), write bf16 [4096][1024]
#pragma unroll
  for (int r = 0; r < 8; ++r) {
    const int m = q0 + r + hfh;
    const float g = gate[(size_t)(b * SEQ + m) * NUM_HEADS + h];
    const float inv = (1.f / (1.f + __expf(-g))) / lrow[r];
#pragma unroll
    for (int t = 0; t < 4; ++t)
      attn[(size_t)(b * SEQ + m) * H_DIM + h * HEAD_DIM + t * 16 + nn] =
          (bf16)(o[t][r] * inv);
  }
}

// ---------------------------------------------------------------- kernel 6
__global__ void __launch_bounds__(128)
gemm_out_kernel(const bf16* __restrict__ A, const bf16* __restrict__ W,
                const float* __restrict__ resid, float* __restrict__ out) {
  const int wave = threadIdx.x >> 5;
  const int lane = threadIdx.x & 31;
  const int row0 = (blockIdx.y * 4 + wave) * 16;
  const int n0   = blockIdx.x * 64;
  v8f acc[4] = {};
  gemm_16x64(A + (size_t)row0 * H_DIM, W + (size_t)n0 * H_DIM, lane, acc);

  const int nn = lane & 15;
  const int hf = lane >> 4;
#pragma unroll
  for (int t = 0; t < 4; ++t) {
    const int n = n0 + t * 16 + nn;
#pragma unroll
    for (int r = 0; r < 8; ++r) {
      const int m = row0 + r + 8 * hf;
      out[(size_t)m * H_DIM + n] = acc[t][r] + resid[(size_t)m * H_DIM + n];
    }
  }
}

// ---------------------------------------------------------------- launch
static constexpr size_t alup(size_t x) { return (x + 255) & ~(size_t)255; }
static constexpr size_t OFF_XN   = 0;
static constexpr size_t OFF_WCAT = alup(OFF_XN   + (size_t)ROWS * H_DIM * 2);
static constexpr size_t OFF_OW   = alup(OFF_WCAT + (size_t)E_PAD * H_DIM * 2);
static constexpr size_t OFF_QK   = alup(OFF_OW   + (size_t)H_DIM * H_DIM * 2);
static constexpr size_t OFF_VT   = alup(OFF_QK   + (size_t)ROWS * 2 * H_DIM * 2);
static constexpr size_t OFF_GATE = alup(OFF_VT   + (size_t)BATCH * NUM_HEADS * HEAD_DIM * SEQ * 2);
static constexpr size_t OFF_ATTN = alup(OFF_GATE + (size_t)ROWS * NUM_HEADS * 4);

extern "C" void kernel_launch(void* const* d_in, const int* in_sizes, int n_in,
                              void* d_out, int out_size, void* d_ws, size_t ws_size,
                              hipStream_t stream) {
  (void)in_sizes; (void)n_in; (void)out_size; (void)ws_size;
  const float* x    = (const float*)d_in[0];
  const float* pw   = (const float*)d_in[1];
  const float* qkvw = (const float*)d_in[2];
  const float* gw   = (const float*)d_in[3];
  const float* ow   = (const float*)d_in[4];
  const float* qnw  = (const float*)d_in[5];
  const float* knw  = (const float*)d_in[6];
  float* out = (float*)d_out;

  char* ws = (char*)d_ws;
  bf16*  xn   = (bf16*)(ws + OFF_XN);
  bf16*  wcat = (bf16*)(ws + OFF_WCAT);
  bf16*  owb  = (bf16*)(ws + OFF_OW);
  bf16*  qkb  = (bf16*)(ws + OFF_QK);
  bf16*  vT   = (bf16*)(ws + OFF_VT);
  float* gate = (float*)(ws + OFF_GATE);
  bf16*  attn = (bf16*)(ws + OFF_ATTN);

  convert_weights_kernel<<<4096, 256, 0, stream>>>(qkvw, gw, ow, wcat, owb);
  prenorm_kernel<<<ROWS, 256, 0, stream>>>(x, pw, xn);
  gemm_qkvgate_kernel<<<dim3(E_PAD / 64, ROWS / 64), 128, 0, stream>>>(
      xn, wcat, qkb, vT, gate);
  qknorm_kernel<<<ROWS, 256, 0, stream>>>(qkb, qnw, knw);
  attn_kernel<<<(BATCH * NUM_HEADS * (SEQ / 16)) / 4, 128, 0, stream>>>(
      qkb, vT, gate, attn);
  gemm_out_kernel<<<dim3(H_DIM / 64, ROWS / 64), 128, 0, stream>>>(
      attn, owb, x, out);
}
